// LSTMDecoder_72945724555373
// MI455X (gfx1250) — compile-verified, matrix operand order checked by reading, not run on hardware
//
#include <hip/hip_runtime.h>
#include <math.h>

// ---------------------------------------------------------------------------
// LSTM decoder with Bahdanau attention, CDNA5 (gfx1250) implementation.
// GEMMs run on v_wmma_f32_16x16x32_bf16 (f32 accumulate); the memory-bound
// attention passes stream bf16 copies of enc_proj / encoder_outputs to halve
// HBM traffic (the binding resource at 23.3 TB/s).
// ---------------------------------------------------------------------------

typedef __bf16 bf16_t;
typedef __bf16 v8bf   __attribute__((ext_vector_type(8)));
typedef __bf16 v16bf  __attribute__((ext_vector_type(16)));
typedef float  v8f    __attribute__((ext_vector_type(8)));

#define B_   512   // batch
#define TH_  512   // history length
#define HE_  256   // encoder hidden
#define H_   256   // decoder hidden
#define A_   256   // attention dim
#define TP_  32    // prediction steps
#define G_   1024  // 4*H gates

// ---------------- WMMA fragment helpers (wave32) ---------------------------
// Row-major [rows x K] bf16 source.  A-frag (M x 32) and B-frag (N x 32,
// i.e. W stored [N x K]) share this layout:
//   lane<16 : row = row0+lane,    K chunks {k0..k0+7}  and {k0+16..k0+23}
//   lane>=16: row = row0+lane-16, K chunks {k0+8..k0+15} and {k0+24..k0+31}
__device__ __forceinline__ v16bf load_frag_rm(const bf16_t* base, int ld, int row0, int k0) {
  int lane = threadIdx.x & 31;
  int r  = row0 + (lane & 15);
  int kk = k0 + ((lane >> 4) << 3);
  const bf16_t* p = base + (size_t)r * ld + kk;
  v8bf lo = *(const v8bf*)(p);
  v8bf hi = *(const v8bf*)(p + 16);
  v16bf f;
#pragma unroll
  for (int i = 0; i < 8; ++i) { f[i] = lo[i]; f[8 + i] = hi[i]; }
  return f;
}

// C/D 16x16 f32 layout: VGPR v -> M=v (lanes 0-15) / M=8+v (lanes 16-31),
// lane%16 -> N.
template <typename OutT>
__device__ __forceinline__ void store_tile(OutT* C, int ld, int m0, int ncol, v8f acc) {
  int lane = threadIdx.x & 31;
  int n  = lane & 15;
  int mb = m0 + ((lane >> 4) << 3);
#pragma unroll
  for (int v = 0; v < 8; ++v)
    C[(size_t)(mb + v) * ld + ncol + n] = (OutT)acc[v];
}

// C[M,N] (f32 or bf16) = A[M,K](bf16) @ B[N,K](bf16)^T.
// Block = 128 threads = 4 waves; wave computes a 16x64 tile; block tile 64x64.
template <typename OutT>
__global__ void gemm_bf16_wmma(const bf16_t* __restrict__ A, const bf16_t* __restrict__ Bw,
                               OutT* __restrict__ C, int M, int N, int K) {
  int wave = threadIdx.x >> 5;
  int m0 = blockIdx.y * 64 + wave * 16;
  int n0 = blockIdx.x * 64;
  if (m0 >= M) return;
  v8f acc0 = {}, acc1 = {}, acc2 = {}, acc3 = {};
  for (int k0 = 0; k0 < K; k0 += 32) {
    v16bf a  = load_frag_rm(A,  K, m0,      k0);
    v16bf b0 = load_frag_rm(Bw, K, n0,      k0);
    v16bf b1 = load_frag_rm(Bw, K, n0 + 16, k0);
    v16bf b2 = load_frag_rm(Bw, K, n0 + 32, k0);
    v16bf b3 = load_frag_rm(Bw, K, n0 + 48, k0);
    acc0 = __builtin_amdgcn_wmma_f32_16x16x32_bf16(false, a, false, b0, (short)0, acc0, false, false);
    acc1 = __builtin_amdgcn_wmma_f32_16x16x32_bf16(false, a, false, b1, (short)0, acc1, false, false);
    acc2 = __builtin_amdgcn_wmma_f32_16x16x32_bf16(false, a, false, b2, (short)0, acc2, false, false);
    acc3 = __builtin_amdgcn_wmma_f32_16x16x32_bf16(false, a, false, b3, (short)0, acc3, false, false);
  }
  store_tile(C, N, m0, n0,      acc0);
  store_tile(C, N, m0, n0 + 16, acc1);
  store_tile(C, N, m0, n0 + 32, acc2);
  store_tile(C, N, m0, n0 + 48, acc3);
}

// ---------------- small helpers --------------------------------------------
__device__ __forceinline__ float wave_sum(float s) {
#pragma unroll
  for (int o = 16; o > 0; o >>= 1) s += __shfl_xor(s, o, 32);
  return s;
}
__device__ __forceinline__ float sigmoidf_(float x) { return 1.0f / (1.0f + expf(-x)); }

// ---------------- one-time prep kernels ------------------------------------
__global__ void conv_bf16_k(const float* __restrict__ in, bf16_t* __restrict__ out, size_t n) {
  size_t i = (size_t)blockIdx.x * 256 + threadIdx.x;
  size_t st = (size_t)gridDim.x * 256;
  for (; i < n; i += st) out[i] = (bf16_t)in[i];
}

// in [R,C] f32 -> out [C,R] bf16
__global__ void transpose_conv_k(const float* __restrict__ in, bf16_t* __restrict__ out, int R, int C) {
  int idx = blockIdx.x * 256 + threadIdx.x;
  if (idx >= R * C) return;
  int r = idx / C, c = idx - r * C;
  out[(size_t)c * R + r] = (bf16_t)in[idx];
}

// out [1024,512] = [ W_ih[:, off0 : off0+256] | W_hh ] as bf16
__global__ void build_wcat_k(const float* __restrict__ Wih, int ih_ld, int off0,
                             const float* __restrict__ Whh, bf16_t* __restrict__ out) {
  int idx = blockIdx.x * 256 + threadIdx.x;
  if (idx >= G_ * 512) return;
  int n = idx >> 9, k = idx & 511;
  float v = (k < 256) ? Wih[(size_t)n * ih_ld + off0 + k] : Whh[(size_t)n * 256 + (k - 256)];
  out[idx] = (bf16_t)v;
}

__global__ void init_state_k(const float* __restrict__ ehid, const float* __restrict__ ecell,
                             bf16_t* __restrict__ X0, bf16_t* __restrict__ X1,
                             bf16_t* __restrict__ htop, float* __restrict__ c0,
                             float* __restrict__ c1, float* __restrict__ dec_in) {
  int b = blockIdx.x, j = threadIdx.x;
  float h0 = ehid[(size_t)b * H_ + j];
  float h1 = ehid[(size_t)(B_ + b) * H_ + j];
  X0[(size_t)b * 512 + 256 + j] = (bf16_t)h0;   // X0 = [ctx | h0_prev]
  X1[(size_t)b * 512 + 256 + j] = (bf16_t)h1;   // X1 = [h0_new | h1_prev]
  htop[(size_t)b * H_ + j]      = (bf16_t)h1;
  c0[(size_t)b * H_ + j] = ecell[(size_t)b * H_ + j];
  c1[(size_t)b * H_ + j] = ecell[(size_t)(B_ + b) * H_ + j];
  if (j == 0) dec_in[b] = 0.0f;
}

// ---------------- per-step kernels -----------------------------------------
// score[b,t] = sum_a tanh(enc_proj[b,t,a] + q[b,a]) * v[a]; one wave per row.
__global__ void score_k(const bf16_t* __restrict__ enc_proj, const float* __restrict__ q,
                        const float* __restrict__ vatt, float* __restrict__ score) {
  int row  = blockIdx.x * 8 + (threadIdx.x >> 5);  // row = b*TH + t
  int lane = threadIdx.x & 31;
  int b = row >> 9;
  const bf16_t* ep = enc_proj + (size_t)row * A_ + lane * 8;
  const float*  qb = q + (size_t)b * A_ + lane * 8;
  const float*  vv = vatt + lane * 8;
  v8bf e = *(const v8bf*)ep;
  float s = 0.0f;
#pragma unroll
  for (int i = 0; i < 8; ++i) s += tanhf((float)e[i] + qb[i]) * vv[i];
  s = wave_sum(s);
  if (lane == 0) score[row] = s;
}

// softmax over T=512 per batch row; writes probabilities into d_out attns slab.
__global__ void softmax_k(const float* __restrict__ score, float* __restrict__ attn_out) {
  __shared__ float red[256];
  int b = blockIdx.x, tid = threadIdx.x;
  const float* s = score + (size_t)b * TH_;
  float m = fmaxf(s[tid], s[tid + 256]);
  red[tid] = m; __syncthreads();
  for (int st = 128; st > 0; st >>= 1) { if (tid < st) red[tid] = fmaxf(red[tid], red[tid + st]); __syncthreads(); }
  m = red[0]; __syncthreads();
  float e0 = expf(s[tid] - m), e1 = expf(s[tid + 256] - m);
  red[tid] = e0 + e1; __syncthreads();
  for (int st = 128; st > 0; st >>= 1) { if (tid < st) red[tid] += red[tid + st]; __syncthreads(); }
  float inv = 1.0f / red[0];
  attn_out[(size_t)b * TH_ + tid]       = e0 * inv;
  attn_out[(size_t)b * TH_ + tid + 256] = e1 * inv;
}

// ctx[b,e] = sum_t attn[b,t] * enc[b,t,e]; writes bf16 copies into X0/X2.
__global__ void ctx_k(const float* __restrict__ attn, const bf16_t* __restrict__ enc,
                      bf16_t* __restrict__ X0, bf16_t* __restrict__ X2) {
  __shared__ float w[TH_];
  int b = blockIdx.x, e = threadIdx.x;
  for (int t = threadIdx.x; t < TH_; t += 256) w[t] = attn[(size_t)b * TH_ + t];
  __syncthreads();
  const bf16_t* eb = enc + (size_t)b * TH_ * HE_ + e;
  float s = 0.0f;
  for (int t = 0; t < TH_; ++t) s += w[t] * (float)eb[(size_t)t * HE_];
  X0[(size_t)b * 512 + e]       = (bf16_t)s;   // lstm0 input cols 0..255
  X2[(size_t)b * 512 + 256 + e] = (bf16_t)s;   // fc input cols 256..511
}

// LSTM layer-0 gate nonlinearity; folds dec_in * W_ih0[:,0] and biases.
__global__ void lstm_update0_k(const float* __restrict__ g, const float* __restrict__ Wih0,
                               const float* __restrict__ bih, const float* __restrict__ bhh,
                               const float* __restrict__ dec_in, float* __restrict__ c,
                               bf16_t* __restrict__ X0, bf16_t* __restrict__ X1) {
  int b = blockIdx.x, j = threadIdx.x;
  float d = dec_in[b];
  const float* gb = g + (size_t)b * G_;
  float gi = gb[j]       + d * Wih0[(size_t)(j)       * 257] + bih[j]       + bhh[j];
  float gf = gb[256 + j] + d * Wih0[(size_t)(256 + j) * 257] + bih[256 + j] + bhh[256 + j];
  float gg = gb[512 + j] + d * Wih0[(size_t)(512 + j) * 257] + bih[512 + j] + bhh[512 + j];
  float go = gb[768 + j] + d * Wih0[(size_t)(768 + j) * 257] + bih[768 + j] + bhh[768 + j];
  float cn = sigmoidf_(gf) * c[(size_t)b * H_ + j] + sigmoidf_(gi) * tanhf(gg);
  float hn = sigmoidf_(go) * tanhf(cn);
  c[(size_t)b * H_ + j] = cn;
  X0[(size_t)b * 512 + 256 + j] = (bf16_t)hn;  // h0 for next step's lstm0
  X1[(size_t)b * 512 + j]       = (bf16_t)hn;  // lstm1 input
}

// LSTM layer-1 gate nonlinearity.
__global__ void lstm_update1_k(const float* __restrict__ g, const float* __restrict__ bih,
                               const float* __restrict__ bhh, float* __restrict__ c,
                               bf16_t* __restrict__ X1, bf16_t* __restrict__ X2,
                               bf16_t* __restrict__ htop) {
  int b = blockIdx.x, j = threadIdx.x;
  const float* gb = g + (size_t)b * G_;
  float gi = gb[j]       + bih[j]       + bhh[j];
  float gf = gb[256 + j] + bih[256 + j] + bhh[256 + j];
  float gg = gb[512 + j] + bih[512 + j] + bhh[512 + j];
  float go = gb[768 + j] + bih[768 + j] + bhh[768 + j];
  float cn = sigmoidf_(gf) * c[(size_t)b * H_ + j] + sigmoidf_(gi) * tanhf(gg);
  float hn = sigmoidf_(go) * tanhf(cn);
  c[(size_t)b * H_ + j] = cn;
  X1[(size_t)b * 512 + 256 + j] = (bf16_t)hn;  // h1 for next step's lstm1
  X2[(size_t)b * 512 + j]       = (bf16_t)hn;  // fc input cols 0..255
  htop[(size_t)b * H_ + j]      = (bf16_t)hn;  // next step's attention query
}

// out[b] = relu(gf[b,:] + b_fc1) . W_fc2 + b_fc2 ; one wave per batch row.
__global__ void fc_tail_k(const float* __restrict__ gf, const float* __restrict__ bfc1,
                          const float* __restrict__ Wfc2, const float* __restrict__ bfc2,
                          float* __restrict__ out, float* __restrict__ dec_in, int step) {
  int wave = threadIdx.x >> 5, lane = threadIdx.x & 31;
  int b = blockIdx.x * 8 + wave;
  const float* g = gf + (size_t)b * H_;
  float acc = 0.0f;
  for (int j = lane; j < H_; j += 32) acc += fmaxf(g[j] + bfc1[j], 0.0f) * Wfc2[j];
  acc = wave_sum(acc);
  if (lane == 0) {
    float r = acc + bfc2[0];
    out[(size_t)b * TP_ + step] = r;  // outputs [B, T_pred, 1]
    dec_in[b] = r;
  }
}

// ---------------------------------------------------------------------------
extern "C" void kernel_launch(void* const* d_in, const int* in_sizes, int n_in,
                              void* d_out, int out_size, void* d_ws, size_t ws_size,
                              hipStream_t stream) {
  (void)in_sizes; (void)n_in; (void)out_size; (void)ws_size;

  const float* enc   = (const float*)d_in[1];   // [B,TH,HE]
  const float* ehid  = (const float*)d_in[2];   // [2,B,H]
  const float* ecell = (const float*)d_in[3];   // [2,B,H]
  const float* Wae   = (const float*)d_in[4];   // [HE,A]
  const float* Wad   = (const float*)d_in[5];   // [H,A]
  const float* vatt  = (const float*)d_in[6];   // [A]
  const float* Wih0  = (const float*)d_in[7];   // [1024,257]
  const float* Whh0  = (const float*)d_in[8];   // [1024,256]
  const float* bih0  = (const float*)d_in[9];
  const float* bhh0  = (const float*)d_in[10];
  const float* Wih1  = (const float*)d_in[11];  // [1024,256]
  const float* Whh1  = (const float*)d_in[12];  // [1024,256]
  const float* bih1  = (const float*)d_in[13];
  const float* bhh1  = (const float*)d_in[14];
  const float* Wfc1  = (const float*)d_in[15];  // [256,512]
  const float* bfc1  = (const float*)d_in[16];
  const float* Wfc2  = (const float*)d_in[17];  // [1,256]
  const float* bfc2  = (const float*)d_in[18];

  // -------- workspace carve ------------------------------------------------
  char* ws = (char*)d_ws;
  size_t off = 0;
  auto carve = [&](size_t bytes) -> char* {
    char* p = ws + off; off = (off + bytes + 255) & ~(size_t)255; return p;
  };
  const size_t BT = (size_t)B_ * TH_;
  bf16_t* enc_bf   = (bf16_t*)carve(BT * HE_ * sizeof(bf16_t));   // 128 MB
  bf16_t* enc_proj = (bf16_t*)carve(BT * A_  * sizeof(bf16_t));   // 128 MB
  bf16_t* Wenc_t   = (bf16_t*)carve((size_t)A_ * HE_ * 2);
  bf16_t* Wdec_t   = (bf16_t*)carve((size_t)A_ * H_  * 2);
  bf16_t* Wcat0    = (bf16_t*)carve((size_t)G_ * 512 * 2);
  bf16_t* Wcat1    = (bf16_t*)carve((size_t)G_ * 512 * 2);
  bf16_t* Wfc1b    = (bf16_t*)carve((size_t)H_ * 512 * 2);
  bf16_t* htop     = (bf16_t*)carve((size_t)B_ * H_ * 2);
  bf16_t* X0       = (bf16_t*)carve((size_t)B_ * 512 * 2);
  bf16_t* X1       = (bf16_t*)carve((size_t)B_ * 512 * 2);
  bf16_t* X2       = (bf16_t*)carve((size_t)B_ * 512 * 2);
  float*  q        = (float*)carve((size_t)B_ * A_ * 4);
  float*  score    = (float*)carve(BT * 4);
  float*  g0       = (float*)carve((size_t)B_ * G_ * 4);
  float*  g1       = (float*)carve((size_t)B_ * G_ * 4);
  float*  gfc      = (float*)carve((size_t)B_ * H_ * 4);
  float*  c0       = (float*)carve((size_t)B_ * H_ * 4);
  float*  c1       = (float*)carve((size_t)B_ * H_ * 4);
  float*  dec_in   = (float*)carve((size_t)B_ * 4);

  float* outs      = (float*)d_out;                  // [B, T_pred]
  float* attn_base = outs + (size_t)B_ * TP_;        // [T_pred, B, TH]

  dim3 gblk(128, 1, 1);

  // -------- one-time prep --------------------------------------------------
  conv_bf16_k<<<4096, 256, 0, stream>>>(enc, enc_bf, BT * HE_);
  transpose_conv_k<<<(HE_ * A_ + 255) / 256, 256, 0, stream>>>(Wae, Wenc_t, HE_, A_);
  transpose_conv_k<<<(H_  * A_ + 255) / 256, 256, 0, stream>>>(Wad, Wdec_t, H_, A_);
  build_wcat_k<<<(G_ * 512 + 255) / 256, 256, 0, stream>>>(Wih0, 257, 1, Whh0, Wcat0);
  build_wcat_k<<<(G_ * 512 + 255) / 256, 256, 0, stream>>>(Wih1, 256, 0, Whh1, Wcat1);
  conv_bf16_k<<<512, 256, 0, stream>>>(Wfc1, Wfc1b, (size_t)H_ * 512);
  init_state_k<<<B_, 256, 0, stream>>>(ehid, ecell, X0, X1, htop, c0, c1, dec_in);

  // enc_proj[b,t,:] = enc[b,t,:] @ W_att_enc   (M = B*TH, N = A, K = HE)
  gemm_bf16_wmma<bf16_t><<<dim3(A_ / 64, (unsigned)(BT / 64)), gblk, 0, stream>>>(
      enc_bf, Wenc_t, enc_proj, (int)BT, A_, HE_);

  // -------- decode steps ---------------------------------------------------
  for (int s = 0; s < TP_; ++s) {
    float* attn_s = attn_base + (size_t)s * B_ * TH_;

    // q = h1 @ W_att_dec
    gemm_bf16_wmma<float><<<dim3(A_ / 64, B_ / 64), gblk, 0, stream>>>(
        htop, Wdec_t, q, B_, A_, H_);
    score_k<<<(unsigned)(BT / 8), 256, 0, stream>>>(enc_proj, q, vatt, score);
    softmax_k<<<B_, 256, 0, stream>>>(score, attn_s);
    ctx_k<<<B_, 256, 0, stream>>>(attn_s, enc_bf, X0, X2);

    // gates0 = [ctx|h0] @ [W_ih0'|W_hh0]^T
    gemm_bf16_wmma<float><<<dim3(G_ / 64, B_ / 64), gblk, 0, stream>>>(
        X0, Wcat0, g0, B_, G_, 512);
    lstm_update0_k<<<B_, 256, 0, stream>>>(g0, Wih0, bih0, bhh0, dec_in, c0, X0, X1);

    // gates1 = [h0|h1] @ [W_ih1|W_hh1]^T
    gemm_bf16_wmma<float><<<dim3(G_ / 64, B_ / 64), gblk, 0, stream>>>(
        X1, Wcat1, g1, B_, G_, 512);
    lstm_update1_k<<<B_, 256, 0, stream>>>(g1, bih1, bhh1, c1, X1, X2, htop);

    // fc1 = [h1|ctx] @ W_fc1^T
    gemm_bf16_wmma<float><<<dim3(H_ / 64, B_ / 64), gblk, 0, stream>>>(
        X2, Wfc1b, gfc, B_, H_, 512);
    fc_tail_k<<<B_ / 8, 256, 0, stream>>>(gfc, bfc1, Wfc2, bfc2, outs, dec_in, s);
  }
}
